// Attention_36034775613597
// MI455X (gfx1250) — matrix operand order repaired and checked by reading, not compiled, not run on hardware
//
#include <hip/hip_runtime.h>
#include <hip/hip_bf16.h>

// ---------- types ----------
typedef __bf16 bf16;
typedef __attribute__((ext_vector_type(4)))  bf16  bf16x4;
typedef __attribute__((ext_vector_type(8)))  bf16  bf16x8;
typedef __attribute__((ext_vector_type(16))) bf16  bf16x16;
typedef __attribute__((ext_vector_type(8)))  float v8f;

union FragAB { bf16x16 v; bf16x8 h[2]; };

// ---------- problem constants ----------
constexpr int S    = 2048;
constexpr int H    = 4096;
constexpr int NH   = 32;
constexpr int NKV  = 8;
constexpr int HD   = 128;
constexpr int GROUP = NH / NKV;                 // 4
constexpr int QKVN = H + 2 * NKV * HD;          // 6144
constexpr float SCALE = 0.08838834764831845f;

// ---------- CDNA5 async global->LDS copy (ASYNCcnt path) ----------
// GLOBAL_LOAD_ASYNC_TO_LDS_B128: per-lane 16B, memory -> LDS, no VGPR data.
__device__ __forceinline__ void async_ld_lds_b128(void* ldsPtr, const void* gptr) {
  uint32_t l = (uint32_t)(uintptr_t)ldsPtr;     // addr[31:0] == LDS byte offset
  asm volatile("global_load_async_to_lds_b128 %0, %1, off"
               :: "v"(l), "v"(gptr)
               : "memory");
}
// Async loads complete in order: waiting <=4 releases the previous slab's 4
// loads while the just-issued next slab's 4 remain in flight.
__device__ __forceinline__ void wait_async_le4() {
  asm volatile("s_wait_asynccnt 0x4" ::: "memory");
}
__device__ __forceinline__ void wait_async0() {
  asm volatile("s_wait_asynccnt 0x0" ::: "memory");
}

// =====================================================================
// Kernel 1: fp32 -> bf16 cast (vectorized, grid-stride)
// =====================================================================
__global__ __launch_bounds__(256) void cast_f32_bf16(const float* __restrict__ in,
                                                     bf16* __restrict__ out, long n4) {
  long i = (long)blockIdx.x * blockDim.x + threadIdx.x;
  long stride = (long)gridDim.x * blockDim.x;
  for (; i < n4; i += stride) {
    float4 v = ((const float4*)in)[i];
    bf16x4 o;
    o.x = (bf16)v.x; o.y = (bf16)v.y; o.z = (bf16)v.z; o.w = (bf16)v.w;
    ((bf16x4*)out)[i] = o;
  }
}

// =====================================================================
// Kernel 2: C[M,N] = A[M,K] * B[N,K]^T   (bf16 in, f32 acc)
// Block: 256 thr (8 waves), tile 128x128, K-step 32.
// Double-buffered LDS fed by GLOBAL_LOAD_ASYNC_TO_LDS_B128: slab k+1's
// copy is in flight behind slab k's WMMAs (in-order ASYNCcnt retirement).
// Wave grid 2(M) x 4(N); each wave: 4x2 tiles of 16x16.
// =====================================================================
template <bool OUT_BF16>
__global__ __launch_bounds__(256) void gemm_bt_wmma(const bf16* __restrict__ A,
                                                    const bf16* __restrict__ B,
                                                    void* __restrict__ Cout,
                                                    int M, int N, int K) {
  constexpr int LDSTR = 40;                       // padded row stride (elems)
  __shared__ bf16 As[2][128 * LDSTR];
  __shared__ bf16 Bs[2][128 * LDSTR];

  const int tid  = threadIdx.x;
  const int lane = tid & 31;
  const int wv   = tid >> 5;
  const int waveM = wv & 1;                       // 0..1 -> 64 rows each
  const int waveN = wv >> 1;                      // 0..3 -> 32 cols each
  const int lh = lane & 15, ls = lane >> 4;
  const int m0 = blockIdx.y * 128, n0 = blockIdx.x * 128;

  const int lr = tid >> 1;                        // 0..127: staged row
  const int lc = (tid & 1) * 16;                  // 0 / 16: K-half

  v8f acc[4][2];
#pragma unroll
  for (int mt = 0; mt < 4; ++mt)
#pragma unroll
    for (int nt = 0; nt < 2; ++nt) acc[mt][nt] = (v8f){};

  const bf16* agBase = A + (size_t)(m0 + lr) * K + lc;
  const bf16* bgBase = B + (size_t)(n0 + lr) * K + lc;

  auto issueSlab = [&](int buf, int k0) {
    bf16* ad = As[buf] + lr * LDSTR + lc;
    bf16* bd = Bs[buf] + lr * LDSTR + lc;
    async_ld_lds_b128(ad,     agBase + k0);
    async_ld_lds_b128(ad + 8, agBase + k0 + 8);
    async_ld_lds_b128(bd,     bgBase + k0);
    async_ld_lds_b128(bd + 8, bgBase + k0 + 8);
  };

  issueSlab(0, 0);                                // prologue: slab 0 in flight

  for (int k0 = 0; k0 < K; k0 += 32) {
    const int cur = (k0 >> 5) & 1;
    if (k0 + 32 < K) {
      issueSlab(cur ^ 1, k0 + 32);                // overlap next copy w/ compute
      wait_async_le4();                           // slab k landed; k+1 in flight
    } else {
      wait_async0();
    }
    if (k0 + 64 < K) {                            // L2 prefetch two slabs ahead
      __builtin_prefetch(agBase + k0 + 64, 0, 1);
      __builtin_prefetch(bgBase + k0 + 64, 0, 1);
    }
    __syncthreads();                              // all waves' writes visible

    // A fragments: lane lh = row; K split by lane-half {0..7,16..23}/{8..15,24..31}
    FragAB af[4];
#pragma unroll
    for (int mt = 0; mt < 4; ++mt) {
      const bf16* p = As[cur] + (waveM * 64 + mt * 16 + lh) * LDSTR + ls * 8;
      af[mt].h[0] = *(const bf16x8*)p;
      af[mt].h[1] = *(const bf16x8*)(p + 16);
    }
    // B fragments: lane lh = column (row of B); K split by lane-half {0..7,16..23}/{8..15,24..31}
    FragAB bfr[2];
#pragma unroll
    for (int nt = 0; nt < 2; ++nt) {
      const bf16* p = Bs[cur] + (waveN * 32 + nt * 16 + lh) * LDSTR + ls * 8;
      bfr[nt].h[0] = *(const bf16x8*)p;
      bfr[nt].h[1] = *(const bf16x8*)(p + 16);
    }
#pragma unroll
    for (int mt = 0; mt < 4; ++mt)
#pragma unroll
      for (int nt = 0; nt < 2; ++nt)
        acc[mt][nt] = __builtin_amdgcn_wmma_f32_16x16x32_bf16(
            false, af[mt].v, false, bfr[nt].v, (short)0, acc[mt][nt], false, false);

    __syncthreads();                              // readers done before overwrite
  }

  // Epilogue: C layout: VGPR j -> row j + 8*(lane/16); col = lane%16
#pragma unroll
  for (int mt = 0; mt < 4; ++mt)
#pragma unroll
    for (int nt = 0; nt < 2; ++nt)
#pragma unroll
      for (int j = 0; j < 8; ++j) {
        int row = m0 + waveM * 64 + mt * 16 + j + 8 * ls;
        int col = n0 + waveN * 32 + nt * 16 + lh;
        float v = acc[mt][nt][j];
        if (OUT_BF16) ((bf16*)Cout)[(size_t)row * N + col] = (bf16)v;
        else          ((float*)Cout)[(size_t)row * N + col] = v;
      }
}

// =====================================================================
// Kernel 3: RoPE on Q (scaled) and K; transpose V to [kvh][d][s]
// =====================================================================
__global__ __launch_bounds__(128) void rope_split(const bf16* __restrict__ QKV,
                                                  const float* __restrict__ cosT,
                                                  const float* __restrict__ sinT,
                                                  bf16* __restrict__ Qr,
                                                  bf16* __restrict__ Kr,
                                                  bf16* __restrict__ Vt) {
  const int s   = blockIdx.x;
  const int kvh = blockIdx.y;
  const int d   = threadIdx.x;                    // 0..127
  const bf16* base = QKV + (size_t)s * QKVN + kvh * ((GROUP + 2) * HD);
  const float c  = cosT[s * HD + d];
  const float sn = sinT[s * HD + d];
#pragma unroll
  for (int g = 0; g < GROUP; ++g) {
    float x  = (float)base[g * HD + d];
    float xr = (d < 64) ? -(float)base[g * HD + d + 64] : (float)base[g * HD + d - 64];
    Qr[((size_t)(kvh * GROUP + g) * S + s) * HD + d] = (bf16)((x * c + xr * sn) * SCALE);
  }
  {
    float x  = (float)base[GROUP * HD + d];
    float xr = (d < 64) ? -(float)base[GROUP * HD + d + 64]
                        : (float)base[GROUP * HD + d - 64];
    Kr[((size_t)kvh * S + s) * HD + d] = (bf16)(x * c + xr * sn);
  }
  Vt[((size_t)kvh * HD + d) * S + s] = base[(GROUP + 1) * HD + d];
}

// =====================================================================
// Kernel 4: causal GQA flash attention.
// Block: 128 thr (4 waves). blockIdx = (q-tile of 64, head).
// Each wave owns 16 q-rows x 128 dims of O (8 v8f accumulators).
// kv chunks of 32: 8 WMMA for QK^T, 8 WMMA for PV.
// =====================================================================
__global__ __launch_bounds__(128) void attn_fwd(const bf16* __restrict__ Qr,
                                                const bf16* __restrict__ Kr,
                                                const bf16* __restrict__ Vt,
                                                bf16* __restrict__ Ob) {
  __shared__ bf16 Plds[4][16][32];
  const int tid = threadIdx.x, lane = tid & 31, w = tid >> 5;
  const int lh = lane & 15, ls = lane >> 4;
  const int h = blockIdx.y, kvh = h >> 2;         // GROUP = 4
  const int qt = blockIdx.x;
  const int qw = qt * 64 + w * 16;                // wave's q-row base

  // Q fragments (16 rows x 128 dims = 4 A-fragments), resident in VGPRs
  FragAB qf[4];
  const bf16* qrow = Qr + ((size_t)h * S + (qw + lh)) * HD;
#pragma unroll
  for (int kt = 0; kt < 4; ++kt) {
    const int base = kt * 32 + ls * 8;
    qf[kt].h[0] = *(const bf16x8*)(qrow + base);
    qf[kt].h[1] = *(const bf16x8*)(qrow + base + 16);
  }

  v8f accO[8];
#pragma unroll
  for (int dt = 0; dt < 8; ++dt) accO[dt] = (v8f){};
  float mrow[8], lrow[8];
#pragma unroll
  for (int j = 0; j < 8; ++j) { mrow[j] = -3.0e38f; lrow[j] = 0.0f; }

  const int kvEnd = qt * 64 + 64;                 // causal upper bound
  for (int kv0 = 0; kv0 < kvEnd; kv0 += 32) {
    // ---- S = Q K^T : two 16x16 tiles over this 32-col kv chunk ----
    v8f s0 = (v8f){}, s1 = (v8f){};
#pragma unroll
    for (int kt = 0; kt < 4; ++kt) {
      FragAB bk0, bk1;
      const bf16* kb = Kr + ((size_t)kvh * S + kv0 + lh) * HD + kt * 32 + ls * 8;
      bk0.h[0] = *(const bf16x8*)kb;
      bk0.h[1] = *(const bf16x8*)(kb + 16);
      const bf16* kb1 = kb + (size_t)16 * HD;
      bk1.h[0] = *(const bf16x8*)kb1;
      bk1.h[1] = *(const bf16x8*)(kb1 + 16);
      s0 = __builtin_amdgcn_wmma_f32_16x16x32_bf16(false, qf[kt].v, false, bk0.v,
                                                   (short)0, s0, false, false);
      s1 = __builtin_amdgcn_wmma_f32_16x16x32_bf16(false, qf[kt].v, false, bk1.v,
                                                   (short)0, s1, false, false);
    }

    // ---- mask + online softmax (rows live in 16-lane halves) ----
#pragma unroll
    for (int j = 0; j < 8; ++j) {
      const int row = qw + j + 8 * ls;
      const int c0 = kv0 + lh, c1 = c0 + 16;
      float x0 = (c0 <= row) ? s0[j] : -3.0e38f;
      float x1 = (c1 <= row) ? s1[j] : -3.0e38f;
      float mx = fmaxf(x0, x1);
#pragma unroll
      for (int off = 1; off < 16; off <<= 1)
        mx = fmaxf(mx, __shfl_xor(mx, off, 16));
      const float mn = fmaxf(mrow[j], mx);
      const float alpha = __expf(mrow[j] - mn);
      const float p0 = __expf(x0 - mn);
      const float p1 = __expf(x1 - mn);
      float rs = p0 + p1;
#pragma unroll
      for (int off = 1; off < 16; off <<= 1)
        rs += __shfl_xor(rs, off, 16);
      lrow[j] = lrow[j] * alpha + rs;
      mrow[j] = mn;
#pragma unroll
      for (int dt = 0; dt < 8; ++dt) accO[dt][j] = accO[dt][j] * alpha;
      Plds[w][j + 8 * ls][lh]      = (bf16)p0;
      Plds[w][j + 8 * ls][16 + lh] = (bf16)p1;
    }
    asm volatile("" ::: "memory");                // keep ds_store before ds_load

    // ---- repack P (C-layout) -> A-fragment layout via per-wave LDS ----
    FragAB pf;
    pf.h[0] = *(const bf16x8*)&Plds[w][lh][ls * 8];
    pf.h[1] = *(const bf16x8*)&Plds[w][lh][16 + ls * 8];

    // ---- O += P V : V^T layout gives contiguous B-fragment loads ----
#pragma unroll
    for (int dt = 0; dt < 8; ++dt) {
      FragAB vf;
      const bf16* vb = Vt + ((size_t)kvh * HD + dt * 16 + lh) * S + kv0 + ls * 8;
      vf.h[0] = *(const bf16x8*)vb;
      vf.h[1] = *(const bf16x8*)(vb + 16);
      accO[dt] = __builtin_amdgcn_wmma_f32_16x16x32_bf16(false, pf.v, false, vf.v,
                                                         (short)0, accO[dt], false, false);
    }
  }

  // ---- epilogue: O / l, store bf16 to [s][h*128+d] ----
#pragma unroll
  for (int dt = 0; dt < 8; ++dt)
#pragma unroll
    for (int j = 0; j < 8; ++j) {
      const int row = qw + j + 8 * ls;
      const int col = h * HD + dt * 16 + lh;
      Ob[(size_t)row * (NH * HD) + col] = (bf16)(accO[dt][j] / lrow[j]);
    }
}

// =====================================================================
// Host-side launch
// =====================================================================
extern "C" void kernel_launch(void* const* d_in, const int* in_sizes, int n_in,
                              void* d_out, int out_size, void* d_ws, size_t ws_size,
                              hipStream_t stream) {
  const float* hs    = (const float*)d_in[0];     // [S, H]
  const float* wAttn = (const float*)d_in[1];     // [QKVN, H]
  const float* wProj = (const float*)d_in[2];     // [H, H]
  const float* rcos  = (const float*)d_in[3];     // [S, HD]
  const float* rsin  = (const float*)d_in[4];     // [S, HD]
  float* out = (float*)d_out;                     // [S, H]

  char* p = (char*)d_ws;
  bf16* Xb   = (bf16*)p; p += (size_t)S * H * 2;
  bf16* Wab  = (bf16*)p; p += (size_t)QKVN * H * 2;
  bf16* Wpb  = (bf16*)p; p += (size_t)H * H * 2;
  bf16* QKVb = (bf16*)p; p += (size_t)S * QKVN * 2;
  bf16* Qr   = (bf16*)p; p += (size_t)NH * S * HD * 2;
  bf16* Kr   = (bf16*)p; p += (size_t)NKV * S * HD * 2;
  bf16* Vt   = (bf16*)p; p += (size_t)NKV * HD * S * 2;
  bf16* Ob   = (bf16*)p; p += (size_t)S * (NH * HD) * 2;

  // 1) fp32 -> bf16 staging
  cast_f32_bf16<<<2048, 256, 0, stream>>>(hs,    Xb,  (long)S * H / 4);
  cast_f32_bf16<<<4096, 256, 0, stream>>>(wAttn, Wab, (long)QKVN * H / 4);
  cast_f32_bf16<<<4096, 256, 0, stream>>>(wProj, Wpb, (long)H * H / 4);

  // 2) QKV = X * Wattn^T   [S, QKVN] bf16
  gemm_bt_wmma<true><<<dim3(QKVN / 128, S / 128), 256, 0, stream>>>(
      Xb, Wab, (void*)QKVb, S, QKVN, H);

  // 3) RoPE + head split + V transpose
  rope_split<<<dim3(S, NKV), 128, 0, stream>>>(QKVb, rcos, rsin, Qr, Kr, Vt);

  // 4) causal GQA flash attention -> Ob [S, NH*HD] bf16
  attn_fwd<<<dim3(S / 64, NH), 128, 0, stream>>>(Qr, Kr, Vt, Ob);

  // 5) out = O * Wproj^T   f32 straight to d_out
  gemm_bt_wmma<false><<<dim3(H / 128, S / 128), 256, 0, stream>>>(
      Ob, Wpb, (void*)out, S, H, H);
}
